// GCN_77721728188759
// MI455X (gfx1250) — matrix operand order repaired and checked by reading, not compiled, not run on hardware
//
#include <hip/hip_runtime.h>
#include <hip/hip_bf16.h>

typedef __attribute__((ext_vector_type(16))) __bf16 v16bf;
typedef __attribute__((ext_vector_type(8)))  float  v8f;

#define TILE_M 128
#define TILE_N 64
#define BK 32
#define AS_STRIDE 40   // bf16 elems (32 + 8 pad), 80B rows, 16B aligned
#define BS_STRIDE 72   // bf16 elems (64 + 8 pad), 144B rows, 16B aligned

__device__ __forceinline__ __bf16 f2bf(float f) { return (__bf16)f; }

// C[M,N] = A[M,K] @ B[K,N], fp32 in/out, bf16 WMMA inner product.
// Block: 256 thr (8 waves), tile 128x64. Wave: 32x32 (2x2 wmma tiles, 4 accums).
__global__ __launch_bounds__(256) void gemm_bf16_wmma(
    const float* __restrict__ A, const float* __restrict__ B,
    float* __restrict__ C, int M, int N, int K)
{
    __shared__ __bf16 As[TILE_M * AS_STRIDE];
    __shared__ __bf16 Bs[BK * BS_STRIDE];

    const int tid  = threadIdx.x;
    const int lane = tid & 31;
    const int wave = tid >> 5;
    const int wr   = wave >> 1;        // 0..3 : 32-row sub-tile
    const int wc   = wave & 1;         // 0..1 : 32-col sub-strip
    const int m0   = blockIdx.x * TILE_M;
    const int n0   = blockIdx.y * TILE_N;

    // tile-loader thread mappings (8 f32 per thread per chunk)
    const int ar = tid >> 2, ac = (tid & 3) * 8;   // A: rows ar, ar+64; 32 cols
    const int br = tid >> 3, bc = (tid & 7) * 8;   // B: 32 rows x 64 cols

    v8f acc00 = {}, acc01 = {}, acc10 = {}, acc11 = {};

    const int lm = lane & 15;          // A-fragment row within 16
    const int kh = (lane >> 4) * 8;    // A-fragment K-half base

    for (int kb = 0; kb < K; kb += BK) {
        __syncthreads();
        // ---- stage A tile (128 x 32 f32 -> bf16), two row-halves per thread ----
        #pragma unroll
        for (int half = 0; half < 2; ++half) {
            const int r    = ar + half * 64;
            const int grow = m0 + r;
            float v[8];
            if (grow < M) {
                const float* p = A + (size_t)grow * K + kb + ac;
                #pragma unroll
                for (int j = 0; j < 8; ++j) v[j] = p[j];
                if (kb + BK < K)
                    __builtin_prefetch(p + BK, 0, 1);   // global_prefetch_b8
            } else {
                #pragma unroll
                for (int j = 0; j < 8; ++j) v[j] = 0.0f;
            }
            __bf16* d = &As[r * AS_STRIDE + ac];
            #pragma unroll
            for (int j = 0; j < 8; ++j) d[j] = f2bf(v[j]);
        }
        // ---- stage B tile (32 x 64 f32 -> bf16) ----
        {
            const float* p = B + (size_t)(kb + br) * N + n0 + bc;
            __bf16* d = &Bs[br * BS_STRIDE + bc];
            #pragma unroll
            for (int j = 0; j < 8; ++j) d[j] = f2bf(p[j]);
            if (kb + BK < K)
                __builtin_prefetch(p + (size_t)BK * N, 0, 1);
        }
        __syncthreads();

        // ---- A fragments: lane L -> row L&15, K = kh..kh+7 and kh+16..kh+23 ----
        v16bf a0, a1;
        {
            const __bf16* p = &As[(wr * 32 + lm) * AS_STRIDE + kh];
            #pragma unroll
            for (int j = 0; j < 8; ++j) { a0[j] = p[j]; a0[j + 8] = p[j + 16]; }
            const __bf16* q = p + 16 * AS_STRIDE;
            #pragma unroll
            for (int j = 0; j < 8; ++j) { a1[j] = q[j]; a1[j + 8] = q[j + 16]; }
        }
        // ---- B fragments: lane L -> col L&15, K = kh..kh+7 and kh+16..kh+23 ----
        v16bf b0, b1;
        {
            const __bf16* p = &Bs[kh * BS_STRIDE + wc * 32 + lm];
            #pragma unroll
            for (int j = 0; j < 8; ++j) {
                b0[j] = p[j * BS_STRIDE]; b0[j + 8] = p[(j + 16) * BS_STRIDE];
                b1[j] = p[j * BS_STRIDE + 16]; b1[j + 8] = p[(j + 16) * BS_STRIDE + 16];
            }
        }
        acc00 = __builtin_amdgcn_wmma_f32_16x16x32_bf16(false, a0, false, b0,
                                                        (short)0, acc00, false, false);
        acc01 = __builtin_amdgcn_wmma_f32_16x16x32_bf16(false, a0, false, b1,
                                                        (short)0, acc01, false, false);
        acc10 = __builtin_amdgcn_wmma_f32_16x16x32_bf16(false, a1, false, b0,
                                                        (short)0, acc10, false, false);
        acc11 = __builtin_amdgcn_wmma_f32_16x16x32_bf16(false, a1, false, b1,
                                                        (short)0, acc11, false, false);
    }

    // ---- store: VGPR r -> M = r + 8*(lane>=16), N = lane&15 ----
    const int nb = n0 + wc * 32 + (lane & 15);
    const int mb = m0 + wr * 32 + ((lane >> 4) << 3);
    #pragma unroll
    for (int r = 0; r < 8; ++r) {
        const int m = mb + r;
        if (m < M) {
            C[(size_t)m * N + nb]      = acc00[r];
            C[(size_t)m * N + nb + 16] = acc01[r];
        }
        const int m2 = m + 16;
        if (m2 < M) {
            C[(size_t)m2 * N + nb]      = acc10[r];
            C[(size_t)m2 * N + nb + 16] = acc11[r];
        }
    }
}

__global__ void fill_f32(float* p, float v, size_t n) {
    size_t i = (size_t)blockIdx.x * blockDim.x + threadIdx.x;
    size_t stride = (size_t)gridDim.x * blockDim.x;
    for (; i < n; i += stride) p[i] = v;
}

__global__ void degree_accum(const int* __restrict__ dst, float* deg, int E) {
    int i = blockIdx.x * blockDim.x + threadIdx.x;
    if (i < E) atomicAdd(&deg[dst[i]], 1.0f);
}

__global__ void deg_to_dinv(float* d, int n) {
    int i = blockIdx.x * blockDim.x + threadIdx.x;
    if (i < n) { float v = d[i]; d[i] = v > 0.0f ? rsqrtf(v) : 0.0f; }
}

// edge-parallel gather/scale/scatter-add: (F>>2) lanes per edge, float4 each.
__global__ void scatter_edges(const float* __restrict__ h, const float* __restrict__ dinv,
                              const int* __restrict__ src, const int* __restrict__ dst,
                              float* __restrict__ agg, int E, int F, int logTpe)
{
    long long tid = (long long)blockIdx.x * blockDim.x + threadIdx.x;
    int e = (int)(tid >> logTpe);
    if (e >= E) return;
    int f = ((int)tid & ((1 << logTpe) - 1)) << 2;
    int s = src[e], d = dst[e];
    float w = dinv[s] * dinv[d];
    const float4 v = *(const float4*)(h + (size_t)s * F + f);
    float* o = agg + (size_t)d * F + f;
    atomicAdd(o + 0, v.x * w);
    atomicAdd(o + 1, v.y * w);
    atomicAdd(o + 2, v.z * w);
    atomicAdd(o + 3, v.w * w);
}

// out = relu(agg + h*dinv^2 (self-loop) + bias)  -- elementwise, float4
__global__ void epilogue_relu(const float* __restrict__ agg, const float* __restrict__ h,
                              const float* __restrict__ dinv, const float* __restrict__ bias,
                              float* __restrict__ out, int Nn, int F)
{
    size_t i = (size_t)blockIdx.x * blockDim.x + threadIdx.x;
    int q = F >> 2;
    size_t total = (size_t)Nn * q;
    if (i >= total) return;
    int row = (int)(i / q);
    int f   = (int)(i % q) * 4;
    float d2 = dinv[row]; d2 *= d2;
    const float4 a  = *(const float4*)(agg  + (size_t)row * F + f);
    const float4 hv = *(const float4*)(h    + (size_t)row * F + f);
    const float4 b  = *(const float4*)(bias + f);
    float4 r;
    r.x = fmaxf(a.x + hv.x * d2 + b.x, 0.0f);
    r.y = fmaxf(a.y + hv.y * d2 + b.y, 0.0f);
    r.z = fmaxf(a.z + hv.z * d2 + b.z, 0.0f);
    r.w = fmaxf(a.w + hv.w * d2 + b.w, 0.0f);
    *(float4*)(out + (size_t)row * F + f) = r;
}

// v = agg + h*dinv^2 + bias; out = v / max(||v||_2, 1e-12). One wave32 per row (F=128).
__global__ __launch_bounds__(256) void epilogue_norm(
    const float* __restrict__ agg, const float* __restrict__ h,
    const float* __restrict__ dinv, const float* __restrict__ bias,
    float* __restrict__ out, int Nn, int F)
{
    int lane = threadIdx.x & 31;
    int wv   = threadIdx.x >> 5;
    int row  = blockIdx.x * 8 + wv;
    if (row >= Nn) return;
    float d2 = dinv[row]; d2 *= d2;
    int f = lane * 4;
    const float4 a  = *(const float4*)(agg  + (size_t)row * F + f);
    const float4 hv = *(const float4*)(h    + (size_t)row * F + f);
    const float4 b  = *(const float4*)(bias + f);
    float4 v;
    v.x = a.x + hv.x * d2 + b.x;
    v.y = a.y + hv.y * d2 + b.y;
    v.z = a.z + hv.z * d2 + b.z;
    v.w = a.w + hv.w * d2 + b.w;
    float ss = v.x * v.x + v.y * v.y + v.z * v.z + v.w * v.w;
    #pragma unroll
    for (int off = 16; off > 0; off >>= 1) ss += __shfl_xor(ss, off, 32);
    float inv = 1.0f / fmaxf(sqrtf(ss), 1e-12f);
    float4 o = { v.x * inv, v.y * inv, v.z * inv, v.w * inv };
    *(float4*)(out + (size_t)row * F + f) = o;
}

extern "C" void kernel_launch(void* const* d_in, const int* in_sizes, int n_in,
                              void* d_out, int out_size, void* d_ws, size_t ws_size,
                              hipStream_t stream)
{
    const float* x  = (const float*)d_in[0];
    const int*   ei = (const int*)d_in[1];
    const float* W1 = (const float*)d_in[2];
    const float* b1 = (const float*)d_in[3];
    const float* W2 = (const float*)d_in[4];
    const float* b2 = (const float*)d_in[5];

    const int HID = in_sizes[3];           // 256
    const int OUT = in_sizes[5];           // 128
    const int IN  = in_sizes[2] / HID;     // 1536
    const int Nn  = in_sizes[0] / IN;      // 50000
    const int E   = in_sizes[1] / 2;       // 1600000
    const int* src = ei;
    const int* dst = ei + E;

    float* ws   = (float*)d_ws;
    float* dinv = ws;                 ws += Nn;                 // 0.2 MB
    float* h1   = ws;                 ws += (size_t)Nn * HID;   // 51 MB
    float* a1   = ws;                 ws += (size_t)Nn * HID;   // 51 MB (agg1 -> relu'd in place)
    float* h2   = ws;                 ws += (size_t)Nn * OUT;   // 26 MB
    float* agg2 = h1;                 // h1 dead after epilogue1; reuse
    float* out  = (float*)d_out;

    const int T = 256;

    // degrees (self-loop counts as 1) -> dinv
    fill_f32<<<512, T, 0, stream>>>(dinv, 1.0f, (size_t)Nn);
    degree_accum<<<(E + T - 1) / T, T, 0, stream>>>(dst, dinv, E);
    deg_to_dinv<<<(Nn + T - 1) / T, T, 0, stream>>>(dinv, Nn);

    // layer 1: h1 = x @ W1
    dim3 g1((Nn + TILE_M - 1) / TILE_M, HID / TILE_N);
    gemm_bf16_wmma<<<g1, T, 0, stream>>>(x, W1, h1, Nn, HID, IN);

    // aggregate 1
    fill_f32<<<2048, T, 0, stream>>>(a1, 0.0f, (size_t)Nn * HID);
    {
        int logTpe = 31 - __builtin_clz(HID >> 2);            // 6
        long long tot = (long long)E << logTpe;
        scatter_edges<<<(unsigned)((tot + T - 1) / T), T, 0, stream>>>(
            h1, dinv, src, dst, a1, E, HID, logTpe);
    }
    // + self-loop + bias, relu (in place on a1)
    {
        size_t tot = (size_t)Nn * (HID >> 2);
        epilogue_relu<<<(unsigned)((tot + T - 1) / T), T, 0, stream>>>(
            a1, h1, dinv, b1, a1, Nn, HID);
    }

    // layer 2: h2 = a1 @ W2
    dim3 g2((Nn + TILE_M - 1) / TILE_M, OUT / TILE_N);
    gemm_bf16_wmma<<<g2, T, 0, stream>>>(a1, W2, h2, Nn, OUT, HID);

    // aggregate 2
    fill_f32<<<2048, T, 0, stream>>>(agg2, 0.0f, (size_t)Nn * OUT);
    {
        int logTpe = 31 - __builtin_clz(OUT >> 2);            // 5
        long long tot = (long long)E << logTpe;
        scatter_edges<<<(unsigned)((tot + T - 1) / T), T, 0, stream>>>(
            h2, dinv, src, dst, agg2, E, OUT, logTpe);
    }
    // + self-loop + bias, row L2-normalize -> d_out
    epilogue_norm<<<(Nn + 7) / 8, T, 0, stream>>>(agg2, h2, dinv, b2, out, Nn, OUT);
}
